// EnhancedGNN_85899345920520
// MI455X (gfx1250) — compile-verified, matrix operand order checked by reading, not run on hardware
//
#include <hip/hip_runtime.h>
#include <math.h>

// ---------------------------------------------------------------------------
// EnhancedGNN forward for MI455X (gfx1250, wave32).
//   - GEMMs: v_wmma_f32_16x16x32_bf16, 16x32 macro-tile per wave (2 WMMA /
//     K-step, A fragment reused). Weights are pre-transposed + pre-converted
//     to bf16 once per launch so B fragments are contiguous b128 loads.
//   - Edge attention: TDM (tensor_load_to_lds) streams the ea slab into LDS,
//     s_wait_tensorcnt sync; node gathers stay in L2; f32 global atomics
//   - Algebraic refactor: q_i·(ea@We+be) = ea·U[dst] + q_i·be with
//     U[n,c,h] = dot(q[n,h,:], We[c,h*32:]) computed by 4 WMMA GEMMs
// ---------------------------------------------------------------------------

typedef __attribute__((ext_vector_type(16))) __bf16 v16bf;
typedef __attribute__((ext_vector_type(8)))  __bf16 v8bf;
typedef __attribute__((ext_vector_type(8)))  float  v8f;
typedef __attribute__((ext_vector_type(4)))  float  f4;
typedef __attribute__((ext_vector_type(4)))  unsigned int v4u;
typedef __attribute__((ext_vector_type(8)))  int    v8i;
typedef __attribute__((ext_vector_type(4)))  int    v4i;

#define NNODE 50000
#define NEDGE 800000
#define HIDD  128
#define NHEAD 4
#define HDIM  32
#define EDIM  32
#define NLAY  3
#define NGRP  64
#define ECHUNK 256            // edges per LDS slab (256*32*4B = 32 KB)

// Toolchain probe: therock headers ship amd_gfx1250_TDM.h and use the 6-arg
// tensor_load_to_lds builtin; ROCm 7.2 clang-22 declares the 5-arg form.
#if __has_include(<hip/amd_detail/amd_gfx1250_TDM.h>)
#define TDM_6ARG 1
#else
#define TDM_6ARG 0
#endif

// ---------------------------------------------------------------------------
// Weight prep: Bt[n*ldbt + k] = bf16( B[k*brs + n*bcs] )  (transpose+convert)
// ---------------------------------------------------------------------------
__global__ void conv_bf16_t(const float* __restrict__ B, int brs, int bcs,
                            __bf16* __restrict__ Bt, int ldbt, int K, int N)
{
  const int i = blockIdx.x * blockDim.x + threadIdx.x;
  if (i >= K * N) return;
  const int n = i / K, k = i % K;
  Bt[(size_t)n * ldbt + k] = (__bf16)B[(size_t)k * brs + (size_t)n * bcs];
}

// ---------------------------------------------------------------------------
// WMMA GEMM: C[M,N] = act(A[M,K] @ B[K,N] + bias), with B supplied as
// pre-transposed bf16 Bt[N][K] (row stride ldbt). One 16(M)x32(N) macro-tile
// per wave: A fragment loaded/converted once, fed to two WMMAs. Per K-step:
// 4x b128 A loads + 4x b128 B loads + 8 cvt + 2 WMMA.
// Requires M%16==0, N%32==0, K%32==0 (true for every GEMM here).
// ---------------------------------------------------------------------------
__global__ __launch_bounds__(256) void gemm_bias_wmma(
    const float* __restrict__ A, int lda,
    const __bf16* __restrict__ Bt, int ldbt,
    const float* __restrict__ bias,
    float* __restrict__ C, int ldc,
    int M, int K, int N, int relu)
{
  const int lane   = threadIdx.x;          // 0..31
  const int tilesN = N >> 5;               // 32-wide macro columns
  const int total  = (M >> 4) * tilesN;
  const int tile   = blockIdx.x * 8 + threadIdx.y;
  if (tile >= total) return;               // wave-uniform: EXEC stays all-ones
  const int m0   = (tile / tilesN) << 4;
  const int n0   = (tile % tilesN) << 5;
  const int half = lane >> 4;              // K-half select (ISA 7.12.2 layout)
  const int idx  = lane & 15;              // row (A) / col (B,D)

  const float*  arow = A  + (size_t)(m0 + idx) * lda;
  const __bf16* brw0 = Bt + (size_t)(n0 + idx) * ldbt;
  const __bf16* brw1 = Bt + (size_t)(n0 + 16 + idx) * ldbt;

  v8f acc0 = {0.f,0.f,0.f,0.f,0.f,0.f,0.f,0.f};
  v8f acc1 = {0.f,0.f,0.f,0.f,0.f,0.f,0.f,0.f};
  for (int k0 = 0; k0 < K; k0 += 32) {
    // A: lane idx (+0/16) holds row m0+idx, K = half*8 + {0..7} and +16
    const float* ap = arow + k0 + half * 8;
    const f4 a01 = *(const f4*)(ap);
    const f4 a23 = *(const f4*)(ap + 4);
    const f4 a45 = *(const f4*)(ap + 16);
    const f4 a67 = *(const f4*)(ap + 20);
    v16bf a;
#pragma unroll
    for (int i = 0; i < 4; ++i) {
      a[i]      = (__bf16)a01[i];
      a[i + 4]  = (__bf16)a23[i];
      a[i + 8]  = (__bf16)a45[i];
      a[i + 12] = (__bf16)a67[i];
    }
    // B: lane idx (+0/16) holds col n0+idx / n0+16+idx; contiguous in Bt
    const __bf16* bp0 = brw0 + k0 + half * 8;
    const __bf16* bp1 = brw1 + k0 + half * 8;
    const v8bf b0lo = *(const v8bf*)(bp0);
    const v8bf b0hi = *(const v8bf*)(bp0 + 16);
    const v8bf b1lo = *(const v8bf*)(bp1);
    const v8bf b1hi = *(const v8bf*)(bp1 + 16);
    v16bf b0, b1;
#pragma unroll
    for (int i = 0; i < 8; ++i) {
      b0[i] = b0lo[i]; b0[i + 8] = b0hi[i];
      b1[i] = b1lo[i]; b1[i + 8] = b1hi[i];
    }
    acc0 = __builtin_amdgcn_wmma_f32_16x16x32_bf16(
        false, a, false, b0, (short)0, acc0, false, false);
    acc1 = __builtin_amdgcn_wmma_f32_16x16x32_bf16(
        false, a, false, b1, (short)0, acc1, false, false);
  }
  // D layout: VGPR r -> row m0 + half*8 + r, col (n0|n0+16) + idx
  const int   nc0   = n0 + idx;
  const int   nc1   = n0 + 16 + idx;
  const float bb0   = bias ? bias[nc0] : 0.0f;
  const float bb1   = bias ? bias[nc1] : 0.0f;
  const int   mbase = m0 + half * 8;
#pragma unroll
  for (int r = 0; r < 8; ++r) {
    float v0 = acc0[r] + bb0;
    float v1 = acc1[r] + bb1;
    if (relu) { v0 = fmaxf(v0, 0.0f); v1 = fmaxf(v1, 0.0f); }
    C[(size_t)(mbase + r) * ldc + nc0] = v0;
    C[(size_t)(mbase + r) * ldc + nc1] = v1;
  }
}

// ---------------------------------------------------------------------------
// TDM: DMA one contiguous slab (ECHUNK*EDIM f32 = 32KB) of ea into LDS @ 0.
// 1-D tile: data_size=4B, tile_dim0 = ECHUNK*EDIM (fits 16 bits), count=1.
// Issued by one wave; completion via TENSORcnt.
// ---------------------------------------------------------------------------
__device__ inline void tdm_load_ea_slab(const float* gsrc, unsigned n_total_elems)
{
  const unsigned long long ga = (unsigned long long)(const void*)gsrc;
  const unsigned tile0 = ECHUNK * EDIM;          // 8192 elements
  v4u g0;
  g0[0] = 1u;                                     // count=1, user descriptor
  g0[1] = 0u;                                     // lds_addr = 0 (dynamic LDS)
  g0[2] = (unsigned)(ga & 0xFFFFFFFFu);           // global_addr[31:0]
  g0[3] = (unsigned)((ga >> 32) & 0x01FFFFFFu)    // global_addr[56:32]
        | (2u << 30);                             // type = 2 ("image")
  v8i g1;
  g1[0] = (int)(2u << 16);                        // wg_mask=0, data_size=4B
  g1[1] = (int)((n_total_elems & 0xFFFFu) << 16); // tensor_dim0[15:0]
  g1[2] = (int)(((n_total_elems >> 16) & 0xFFFFu) // tensor_dim0[31:16]
        | (1u << 16));                            // tensor_dim1 = 1
  g1[3] = (int)(tile0 << 16);                     // tensor_dim1 hi=0 | tile_dim0
  g1[4] = 0;                                      // tile_dim1=0, tile_dim2=0
  g1[5] = (int)n_total_elems;                     // tensor_dim0_stride lo
  g1[6] = 0;                                      // stride hi | dim1_stride lo
  g1[7] = 0;
  v4i gz = {0, 0, 0, 0};
#if TDM_6ARG
  v8i gz8 = {0, 0, 0, 0, 0, 0, 0, 0};
  __builtin_amdgcn_tensor_load_to_lds(g0, g1, gz, gz, gz8, 0);
#else
  __builtin_amdgcn_tensor_load_to_lds(g0, g1, gz, gz, 0);
#endif
}

// ---------------------------------------------------------------------------
// Edge attention: block handles ECHUNK contiguous edges per iteration.
// Wave 0 TDM-loads the ea slab to LDS, waits TENSORcnt, barrier; then each
// of the 8 waves processes 32 edges:
//   alpha[h] = scale * ( q[dst]·(k[src]+be) + ea·U[dst] )  per head (shfl_xor)
//   softmax over HEAD axis, agg[dst] += v[src]*alpha via f32 atomics.
// ---------------------------------------------------------------------------
__global__ __launch_bounds__(256) void edge_attn_kernel(
    const int* __restrict__ src, const int* __restrict__ dst,
    const float* __restrict__ q, const float* __restrict__ k,
    const float* __restrict__ v, const float* __restrict__ U,
    const float* __restrict__ ea, const float* __restrict__ be_l,
    float* __restrict__ agg, int E)
{
  extern __shared__ float lds_ea[];              // ECHUNK*EDIM floats @ LDS 0
  const int lane = threadIdx.x & 31;
  const int wave = threadIdx.x >> 5;
  const int numChunks = E / ECHUNK;              // 800000/256 = 3125 exact
  const float scale = 0.17677669529663687f;      // 1/sqrt(32)

  for (int c = blockIdx.x; c < numChunks; c += gridDim.x) {
    if (wave == 0) {
      tdm_load_ea_slab(ea + (size_t)c * ECHUNK * EDIM, (unsigned)E * EDIM);
      __builtin_amdgcn_s_wait_tensorcnt(0);
    }
    __syncthreads();                             // slab visible to all waves

    const int base = c * ECHUNK + wave * 32;
#pragma unroll 1
    for (int i = 0; i < 32; ++i) {
      const int e = base + i;
      const int s = src[e];
      const int d = dst[e];
      if (i + 1 < 32) {                          // prefetch next gather rows
        const int sn = src[e + 1], dn = dst[e + 1];
        __builtin_prefetch(&q[(size_t)dn * HIDD + lane * 4], 0, 0);
        __builtin_prefetch(&k[(size_t)sn * HIDD + lane * 4], 0, 0);
        __builtin_prefetch(&v[(size_t)sn * HIDD + lane * 4], 0, 0);
      }
      const float eac = lds_ea[(wave * 32 + i) * EDIM + lane];  // ds_load
      float al[NHEAD];
#pragma unroll
      for (int h = 0; h < NHEAD; ++h) {
        const int dd = h * HDIM + lane;
        const float qv = q[(size_t)d * HIDD + dd];
        float p = qv * (k[(size_t)s * HIDD + dd] + be_l[dd])  // q·k + q·be
                + eac * U[(size_t)d * HIDD + dd];             // ea·U
#pragma unroll
        for (int off = 16; off; off >>= 1) p += __shfl_xor(p, off, 32);
        al[h] = p * scale;
      }
      float mx = fmaxf(fmaxf(al[0], al[1]), fmaxf(al[2], al[3]));
      float sum = 0.f;
#pragma unroll
      for (int h = 0; h < NHEAD; ++h) { al[h] = __expf(al[h] - mx); sum += al[h]; }
      const float inv = 1.0f / sum;
#pragma unroll
      for (int h = 0; h < NHEAD; ++h) {
        const int dd = h * HDIM + lane;
        const float msg = v[(size_t)s * HIDD + dd] * (al[h] * inv);
        atomicAdd(&agg[(size_t)d * HIDD + dd], msg);
      }
    }
    __syncthreads();                             // slab dead before next DMA
  }
}

// ---------------------------------------------------------------------------
// hout = LayerNorm(out + hin) * g + b       (one wave per node, 4 dims/lane)
// ---------------------------------------------------------------------------
__global__ __launch_bounds__(256) void ln_residual_kernel(
    const float* __restrict__ out, const float* __restrict__ hin,
    const float* __restrict__ g, const float* __restrict__ b,
    float* __restrict__ hout, int Nn)
{
  const int lane = threadIdx.x & 31;
  const int n = blockIdx.x * 8 + (threadIdx.x >> 5);
  if (n >= Nn) return;
  float x[4];
  float s = 0.f;
#pragma unroll
  for (int r = 0; r < 4; ++r) {
    const int d = r * 32 + lane;
    x[r] = out[(size_t)n * HIDD + d] + hin[(size_t)n * HIDD + d];
    s += x[r];
  }
#pragma unroll
  for (int off = 16; off; off >>= 1) s += __shfl_xor(s, off, 32);
  const float mu = s * (1.0f / 128.0f);
  float vs = 0.f;
#pragma unroll
  for (int r = 0; r < 4; ++r) { const float t = x[r] - mu; vs += t * t; }
#pragma unroll
  for (int off = 16; off; off >>= 1) vs += __shfl_xor(vs, off, 32);
  const float rstd = rsqrtf(vs * (1.0f / 128.0f) + 1e-5f);
#pragma unroll
  for (int r = 0; r < 4; ++r) {
    const int d = r * 32 + lane;
    hout[(size_t)n * HIDD + d] = (x[r] - mu) * rstd * g[d] + b[d];
  }
}

// ---------------------------------------------------------------------------
// Pooling: segment sum / max / count over batch ids (wave per node).
// ---------------------------------------------------------------------------
__device__ inline void atomicMaxF(float* addr, float val) {
  unsigned* ua = (unsigned*)addr;
  unsigned old = *ua;
  while (__uint_as_float(old) < val) {
    const unsigned assumed = old;
    old = atomicCAS(ua, assumed, __float_as_uint(val));
    if (old == assumed) break;
  }
}

__global__ __launch_bounds__(256) void pool_kernel(
    const float* __restrict__ h, const int* __restrict__ batch,
    float* __restrict__ ssum, float* __restrict__ smax,
    float* __restrict__ cnt, int Nn)
{
  const int lane = threadIdx.x & 31;
  const int n = blockIdx.x * 8 + (threadIdx.x >> 5);
  if (n >= Nn) return;
  const int g = batch[n];
#pragma unroll
  for (int r = 0; r < 4; ++r) {
    const int d = r * 32 + lane;
    const float v = h[(size_t)n * HIDD + d];
    atomicAdd(&ssum[g * HIDD + d], v);
    atomicMaxF(&smax[g * HIDD + d], v);
  }
  if (lane == 0) atomicAdd(&cnt[g], 1.0f);
}

__global__ void init_smax_kernel(float* smax) {
  const int i = blockIdx.x * blockDim.x + threadIdx.x;
  if (i < NGRP * HIDD) smax[i] = -INFINITY;
}

__global__ void finalize_pool_kernel(
    const float* __restrict__ ssum, const float* __restrict__ smax,
    const float* __restrict__ cnt, float* __restrict__ pooled)
{
  const int i = blockIdx.x * blockDim.x + threadIdx.x;
  if (i >= NGRP * HIDD) return;
  const int g = i >> 7, d = i & 127;
  const float c = fmaxf(cnt[g], 1.0f);
  pooled[g * 384 + d]       = ssum[i] / c;   // mean
  pooled[g * 384 + 128 + d] = smax[i];       // max
  pooled[g * 384 + 256 + d] = ssum[i];       // sum
}

// ---------------------------------------------------------------------------
// Host orchestration
// ---------------------------------------------------------------------------
extern "C" void kernel_launch(void* const* d_in, const int* in_sizes, int n_in,
                              void* d_out, int out_size, void* d_ws, size_t ws_size,
                              hipStream_t stream)
{
  (void)in_sizes; (void)n_in; (void)out_size; (void)ws_size;
  const int Nn = NNODE, E = NEDGE;

  const float* x     = (const float*)d_in[0];
  const int*   eidx  = (const int*)  d_in[1];
  const float* eattr = (const float*)d_in[2];
  const int*   batch = (const int*)  d_in[3];
  const float* W_in  = (const float*)d_in[4];
  const float* b_in  = (const float*)d_in[5];
  const float* Wq    = (const float*)d_in[6];
  const float* bq    = (const float*)d_in[7];
  const float* Wk    = (const float*)d_in[8];
  const float* bk    = (const float*)d_in[9];
  const float* Wv    = (const float*)d_in[10];
  const float* bv    = (const float*)d_in[11];
  const float* We    = (const float*)d_in[12];
  const float* be    = (const float*)d_in[13];
  const float* Wo    = (const float*)d_in[14];
  const float* bo    = (const float*)d_in[15];
  const float* ln_g  = (const float*)d_in[16];
  const float* ln_b  = (const float*)d_in[17];
  const float* W_ep  = (const float*)d_in[18];
  const float* b_ep  = (const float*)d_in[19];
  const float* W1    = (const float*)d_in[20];
  const float* b1    = (const float*)d_in[21];
  const float* W2    = (const float*)d_in[22];
  const float* b2    = (const float*)d_in[23];

  const int* src = eidx;         // edge_index[0]
  const int* dst = eidx + E;     // edge_index[1]

  // Workspace layout: f32 region (~282 MB) then bf16 weight region (~0.6 MB).
  float* ws = (float*)d_ws;
  size_t off = 0;
  float* h      = ws + off; off += (size_t)Nn * HIDD;
  float* q      = ws + off; off += (size_t)Nn * HIDD;
  float* k      = ws + off; off += (size_t)Nn * HIDD;
  float* v      = ws + off; off += (size_t)Nn * HIDD;
  float* U      = ws + off; off += (size_t)Nn * HIDD;
  float* agg    = ws + off; off += (size_t)Nn * HIDD;
  float* outb   = ws + off; off += (size_t)Nn * HIDD;
  float* ea     = ws + off; off += (size_t)E  * EDIM;
  float* ssum   = ws + off; off += (size_t)NGRP * HIDD;
  float* smax   = ws + off; off += (size_t)NGRP * HIDD;
  float* cnt    = ws + off; off += (size_t)NGRP;
  float* pooled = ws + off; off += (size_t)NGRP * 384;
  float* hid1   = ws + off; off += (size_t)NGRP * HIDD;

  __bf16* wsb = (__bf16*)(ws + off);   // 16B-aligned (off is a multiple of 4)
  size_t bo_ = 0;
  __bf16* Wept = wsb + bo_; bo_ += 32 * 32;                    // W_ep^T
  __bf16* Wint = wsb + bo_; bo_ += 128 * 128;                  // W_in^T
  __bf16* Wqt  = wsb + bo_; bo_ += (size_t)NLAY * 128 * 128;   // Wq^T per layer
  __bf16* Wkt  = wsb + bo_; bo_ += (size_t)NLAY * 128 * 128;
  __bf16* Wvt  = wsb + bo_; bo_ += (size_t)NLAY * 128 * 128;
  __bf16* Wot  = wsb + bo_; bo_ += (size_t)NLAY * 128 * 128;
  __bf16* Wet  = wsb + bo_; bo_ += (size_t)NLAY * 32 * 128;    // We (plain cvt)
  __bf16* W1t  = wsb + bo_; bo_ += (size_t)128 * 384;          // W1^T
  __bf16* W2t  = wsb + bo_; bo_ += (size_t)128 * 128;          // W2^T

  float* h_final   = (float*)d_out;                       // [N, HID]
  float* graph_out = (float*)d_out + (size_t)Nn * HIDD;   // [G, OUT]

  const dim3 blkG(32, 8);  // 8 waves, one 16x32 macro-tile each
  auto tilesGrid = [](int M, int N) { return ((M >> 4) * (N >> 5) + 7) / 8; };
  auto cgrid     = [](int KN) { return (KN + 255) / 256; };

  // ---- weight prep (bf16, pre-transposed) -------------------------------
  hipLaunchKernelGGL(conv_bf16_t, dim3(cgrid(32*32)), dim3(256), 0, stream,
                     W_ep, 32, 1, Wept, 32, 32, 32);
  hipLaunchKernelGGL(conv_bf16_t, dim3(cgrid(128*128)), dim3(256), 0, stream,
                     W_in, 128, 1, Wint, 128, 128, 128);
  for (int l = 0; l < NLAY; ++l) {
    hipLaunchKernelGGL(conv_bf16_t, dim3(cgrid(128*128)), dim3(256), 0, stream,
                       Wq + (size_t)l*128*128, 128, 1, Wqt + (size_t)l*128*128, 128, 128, 128);
    hipLaunchKernelGGL(conv_bf16_t, dim3(cgrid(128*128)), dim3(256), 0, stream,
                       Wk + (size_t)l*128*128, 128, 1, Wkt + (size_t)l*128*128, 128, 128, 128);
    hipLaunchKernelGGL(conv_bf16_t, dim3(cgrid(128*128)), dim3(256), 0, stream,
                       Wv + (size_t)l*128*128, 128, 1, Wvt + (size_t)l*128*128, 128, 128, 128);
    hipLaunchKernelGGL(conv_bf16_t, dim3(cgrid(128*128)), dim3(256), 0, stream,
                       Wo + (size_t)l*128*128, 128, 1, Wot + (size_t)l*128*128, 128, 128, 128);
    // Wet[c][d] = bf16(We_l[c][d]) (identity layout; serves all 4 heads)
    hipLaunchKernelGGL(conv_bf16_t, dim3(cgrid(128*32)), dim3(256), 0, stream,
                       We + (size_t)l*EDIM*HIDD, 1, 128, Wet + (size_t)l*32*128, 128, 128, 32);
  }
  hipLaunchKernelGGL(conv_bf16_t, dim3(cgrid(384*128)), dim3(256), 0, stream,
                     W1, 128, 1, W1t, 384, 384, 128);
  hipLaunchKernelGGL(conv_bf16_t, dim3(cgrid(128*128)), dim3(256), 0, stream,
                     W2, 128, 1, W2t, 128, 128, 128);

  // ---- input / edge projections -----------------------------------------
  hipLaunchKernelGGL(gemm_bias_wmma, dim3(tilesGrid(E, EDIM)), blkG, 0, stream,
                     eattr, EDIM, Wept, 32, b_ep, ea, EDIM, E, EDIM, EDIM, 0);
  hipLaunchKernelGGL(gemm_bias_wmma, dim3(tilesGrid(Nn, HIDD)), blkG, 0, stream,
                     x, 128, Wint, 128, b_in, h, HIDD, Nn, 128, HIDD, 0);

  for (int l = 0; l < NLAY; ++l) {
    // q/k/v projections
    hipLaunchKernelGGL(gemm_bias_wmma, dim3(tilesGrid(Nn, HIDD)), blkG, 0, stream,
                       h, HIDD, Wqt + (size_t)l*128*128, 128, bq + l*HIDD, q, HIDD, Nn, HIDD, HIDD, 0);
    hipLaunchKernelGGL(gemm_bias_wmma, dim3(tilesGrid(Nn, HIDD)), blkG, 0, stream,
                       h, HIDD, Wkt + (size_t)l*128*128, 128, bk + l*HIDD, k, HIDD, Nn, HIDD, HIDD, 0);
    hipLaunchKernelGGL(gemm_bias_wmma, dim3(tilesGrid(Nn, HIDD)), blkG, 0, stream,
                       h, HIDD, Wvt + (size_t)l*128*128, 128, bv + l*HIDD, v, HIDD, Nn, HIDD, HIDD, 0);

    // U[n, h*32+c] = sum_d q[n,h*32+d] * We_l[c, h*32+d]
    for (int hh = 0; hh < NHEAD; ++hh) {
      hipLaunchKernelGGL(gemm_bias_wmma, dim3(tilesGrid(Nn, HDIM)), blkG, 0, stream,
                         q + hh * HDIM, HIDD,                     // A = q head slice
                         Wet + (size_t)l*32*128 + hh * HDIM, 128, // Bt = We rows
                         (const float*)nullptr,
                         U + hh * HDIM, HIDD,
                         Nn, HDIM, HDIM, 0);
    }

    // edge attention + aggregation (TDM-staged ea slabs, 32KB dynamic LDS)
    hipMemsetAsync(agg, 0, (size_t)Nn * HIDD * sizeof(float), stream);
    hipLaunchKernelGGL(edge_attn_kernel, dim3(1024), dim3(256),
                       ECHUNK * EDIM * sizeof(float), stream,
                       src, dst, q, k, v, U, ea, be + l * HIDD, agg, E);

    // out = agg @ Wo + bo ; h = LN(out + h)  (last layer writes d_out)
    hipLaunchKernelGGL(gemm_bias_wmma, dim3(tilesGrid(Nn, HIDD)), blkG, 0, stream,
                       agg, HIDD, Wot + (size_t)l*128*128, 128, bo + l*HIDD, outb, HIDD, Nn, HIDD, HIDD, 0);
    float* hout = (l == NLAY - 1) ? h_final : h;
    hipLaunchKernelGGL(ln_residual_kernel, dim3((Nn + 7) / 8), dim3(256), 0, stream,
                       outb, h, ln_g + l * HIDD, ln_b + l * HIDD, hout, Nn);
  }

  // graph pooling (mean/max/sum) over batch ids
  hipMemsetAsync(ssum, 0, (size_t)NGRP * HIDD * sizeof(float), stream);
  hipMemsetAsync(cnt, 0, (size_t)NGRP * sizeof(float), stream);
  hipLaunchKernelGGL(init_smax_kernel, dim3((NGRP * HIDD + 255) / 256), dim3(256), 0, stream,
                     smax);
  hipLaunchKernelGGL(pool_kernel, dim3((Nn + 7) / 8), dim3(256), 0, stream,
                     h_final, batch, ssum, smax, cnt, Nn);
  hipLaunchKernelGGL(finalize_pool_kernel, dim3((NGRP * HIDD + 255) / 256), dim3(256), 0, stream,
                     ssum, smax, cnt, pooled);

  // graph head: relu(pooled @ W1 + b1) @ W2 + b2   (64 rows -> 4 M-tiles)
  hipLaunchKernelGGL(gemm_bias_wmma, dim3(tilesGrid(NGRP, HIDD)), blkG, 0, stream,
                     pooled, 384, W1t, 384, b1, hid1, HIDD, NGRP, 384, HIDD, 1);
  hipLaunchKernelGGL(gemm_bias_wmma, dim3(tilesGrid(NGRP, HIDD)), blkG, 0, stream,
                     hid1, HIDD, W2t, 128, b2, graph_out, HIDD, NGRP, HIDD, HIDD, 0);
}
